// ModifiedGATConv_40862318854644
// MI455X (gfx1250) — matrix operand order repaired and checked by reading, not compiled, not run on hardware
//
#include <hip/hip_runtime.h>
#include <hip/hip_bf16.h>
#include <math.h>

#define NN   30000
#define EE   480000
#define INF_ 128
#define HH   4
#define DD   128
#define HDIM 512
#define NEG_SLOPE 0.2f
#define BN_EPS 1e-5f

typedef __attribute__((ext_vector_type(16))) __bf16 v16bf;
typedef __attribute__((ext_vector_type(8)))  __bf16 v8bf;
typedef __attribute__((ext_vector_type(8)))  float  v8f;

// ---------------- helpers ----------------

__device__ __forceinline__ float warp_reduce_sum(float v) {
#pragma unroll
  for (int off = 16; off > 0; off >>= 1)
    v += __shfl_xor(v, off, 32);
  return v;
}

__device__ __forceinline__ void atomic_max_float(float* addr, float val) {
  if (val >= 0.0f)
    atomicMax((int*)addr, __float_as_int(val));
  else
    atomicMin((unsigned int*)addr, __float_as_uint(val));
}

// 16 contiguous bf16 -> v16bf (two 16B vector loads)
__device__ __forceinline__ v16bf load16(const __bf16* p) {
  v8bf lo = *(const v8bf*)p;
  v8bf hi = *(const v8bf*)(p + 8);
  return __builtin_shufflevector(lo, hi, 0, 1, 2, 3, 4, 5, 6, 7,
                                 8, 9, 10, 11, 12, 13, 14, 15);
}

// two separate 8-element bf16 runs -> v16bf
__device__ __forceinline__ v16bf load8x2(const __bf16* p0, const __bf16* p1) {
  v8bf lo = *(const v8bf*)p0;
  v8bf hi = *(const v8bf*)p1;
  return __builtin_shufflevector(lo, hi, 0, 1, 2, 3, 4, 5, 6, 7,
                                 8, 9, 10, 11, 12, 13, 14, 15);
}

__device__ __forceinline__ void pack8(v16bf& a, int base, float4 u, float4 v) {
  a[base + 0] = (__bf16)u.x; a[base + 1] = (__bf16)u.y;
  a[base + 2] = (__bf16)u.z; a[base + 3] = (__bf16)u.w;
  a[base + 4] = (__bf16)v.x; a[base + 5] = (__bf16)v.y;
  a[base + 6] = (__bf16)v.z; a[base + 7] = (__bf16)v.w;
}

__device__ __forceinline__ v8f bf16_wmma(v16bf a, v16bf b, v8f c) {
  return __builtin_amdgcn_wmma_f32_16x16x32_bf16(false, a, false, b,
                                                 (short)0, c, false, false);
}

__device__ __forceinline__ float edge_score(const float* el, const float* er,
                                            int s, int d, int h) {
  float e = el[s * HH + h] + er[d * HH + h];
  return e > 0.f ? e : NEG_SLOPE * e;
}

// ---------------- prep kernels ----------------

__global__ void k_cvt_feat(const float* __restrict__ f, __bf16* __restrict__ o) {
  int i = blockIdx.x * blockDim.x + threadIdx.x;
  if (i >= NN * INF_ / 4) return;
  float4 v = ((const float4*)f)[i];
  __bf16* p = o + (size_t)i * 4;
  p[0] = (__bf16)v.x; p[1] = (__bf16)v.y;
  p[2] = (__bf16)v.z; p[3] = (__bf16)v.w;
}

// transpose + convert: in [K][Ncols] fp32 -> out [Ncols][K] bf16
__global__ void k_transpose_w(const float* __restrict__ w, __bf16* __restrict__ wt,
                              int K, int Ncols) {
  int i = blockIdx.x * blockDim.x + threadIdx.x;
  if (i >= K * Ncols) return;
  int n = i / K, k = i % K;
  wt[i] = (__bf16)w[(size_t)k * Ncols + n];
}

__global__ void k_init(float* __restrict__ emax, float* __restrict__ denom,
                       float* __restrict__ bnsum, float* __restrict__ bnsq) {
  int i = blockIdx.x * blockDim.x + threadIdx.x;
  if (i < NN * HH) { emax[i] = -INFINITY; denom[i] = 0.f; }
  if (i < HDIM)    { bnsum[i] = 0.f; bnsq[i] = 0.f; }
}

// ---------------- GEMM kernels ----------------

// h = feat @ W_fc (bf16 out) and rst = feat @ W_res + bias (fp32 out).
// One wave = one 16x32 output slab of BOTH matrices. All 5 fragments of a
// K-step are loaded into distinct live values BEFORE any WMMA so the clause
// keeps 10 b128 loads in flight instead of serializing on one register octet.
__global__ void __launch_bounds__(256)
k_gemm_fc_res(const __bf16* __restrict__ featbf, const __bf16* __restrict__ wtfc,
              const __bf16* __restrict__ wtres, const float* __restrict__ bias,
              __bf16* __restrict__ hbf, float* __restrict__ rst) {
  const int wave = (int)((blockIdx.x * blockDim.x + threadIdx.x) >> 5);
  const int lane = threadIdx.x & 31;
  const int nCP = HDIM / 32;                        // 16 column-pairs
  if (wave >= (NN / 16) * nCP) return;              // exact grid; wave-uniform
  const int row0 = (wave / nCP) * 16;
  const int col0 = (wave % nCP) * 32;
  const int laneN = lane & 15, hi = lane >> 4;

  const __bf16* Arow = featbf + (size_t)(row0 + laneN) * INF_;
  const __bf16* Bf0 = wtfc  + (size_t)(col0 + laneN) * INF_;
  const __bf16* Bf1 = wtfc  + (size_t)(col0 + 16 + laneN) * INF_;
  const __bf16* Br0 = wtres + (size_t)(col0 + laneN) * INF_;
  const __bf16* Br1 = wtres + (size_t)(col0 + 16 + laneN) * INF_;

  v8f cf0 = {}, cf1 = {}, cr0 = {}, cr1 = {};
#pragma unroll
  for (int k0 = 0; k0 < INF_; k0 += 32) {
    const int kb = k0 + (hi << 4);
    // issue all loads of this K-step first (distinct destinations)
    v16bf a  = load16(Arow + kb);
    v16bf b0 = load16(Bf0 + kb);
    v16bf b1 = load16(Bf1 + kb);
    v16bf b2 = load16(Br0 + kb);
    v16bf b3 = load16(Br1 + kb);
    // then drain through the matrix unit
    cf0 = bf16_wmma(a, b0, cf0);
    cf1 = bf16_wmma(a, b1, cf1);
    cr0 = bf16_wmma(a, b2, cr0);
    cr1 = bf16_wmma(a, b3, cr1);
  }
  const float bv0 = bias[col0 + laneN];
  const float bv1 = bias[col0 + 16 + laneN];
#pragma unroll
  for (int v = 0; v < 8; ++v) {
    int m = v + (hi << 3);
    size_t base = (size_t)(row0 + m) * HDIM + laneN;
    hbf[base + col0]      = (__bf16)cf0[v];
    hbf[base + col0 + 16] = (__bf16)cf1[v];
    rst[base + col0]      = cr0[v] + bv0;
    rst[base + col0 + 16] = cr1[v] + bv1;
  }
}

// out = relu(x*scale+shift) @ W_out + b_out; BN+ReLU fused into A fragment.
__global__ void __launch_bounds__(256)
k_gemm_out(const float* __restrict__ x, const float* __restrict__ scale,
           const float* __restrict__ shift, const __bf16* __restrict__ wtout,
           const float* __restrict__ bout, float* __restrict__ out) {
  const int wave = (int)((blockIdx.x * blockDim.x + threadIdx.x) >> 5);
  const int lane = threadIdx.x & 31;
  const int nCP = INF_ / 32;                        // 4 column-pairs
  if (wave >= (NN / 16) * nCP) return;              // wave-uniform guard
  const int row0 = (wave / nCP) * 16;
  const int col0 = (wave % nCP) * 32;
  const int laneN = lane & 15, hi = lane >> 4;

  const float*  Arow = x + (size_t)(row0 + laneN) * HDIM;
  const __bf16* B0 = wtout + (size_t)(col0 + laneN) * HDIM;
  const __bf16* B1 = wtout + (size_t)(col0 + 16 + laneN) * HDIM;

  v8f c0 = {}, c1 = {};
#pragma unroll 2
  for (int k0 = 0; k0 < HDIM; k0 += 32) {
    const int kb0 = k0 + (hi << 4);
    const int kb1 = k0 + (hi << 4) + 8;
    const int kb  = k0 + (hi << 4);
    // all loads of this K-step first
    float4 x0 = *(const float4*)(Arow + kb0);
    float4 x1 = *(const float4*)(Arow + kb0 + 4);
    float4 x2 = *(const float4*)(Arow + kb1);
    float4 x3 = *(const float4*)(Arow + kb1 + 4);
    v16bf b0 = load16(B0 + kb);
    v16bf b1 = load16(B1 + kb);
    float4 sc0 = *(const float4*)(scale + kb0);
    float4 sc1 = *(const float4*)(scale + kb0 + 4);
    float4 sc2 = *(const float4*)(scale + kb1);
    float4 sc3 = *(const float4*)(scale + kb1 + 4);
    float4 sh0 = *(const float4*)(shift + kb0);
    float4 sh1 = *(const float4*)(shift + kb0 + 4);
    float4 sh2 = *(const float4*)(shift + kb1);
    float4 sh3 = *(const float4*)(shift + kb1 + 4);
#define BNRELU4(v, s, b) \
    v.x = fmaxf(v.x * s.x + b.x, 0.f); v.y = fmaxf(v.y * s.y + b.y, 0.f); \
    v.z = fmaxf(v.z * s.z + b.z, 0.f); v.w = fmaxf(v.w * s.w + b.w, 0.f);
    BNRELU4(x0, sc0, sh0) BNRELU4(x1, sc1, sh1)
    BNRELU4(x2, sc2, sh2) BNRELU4(x3, sc3, sh3)
#undef BNRELU4
    v16bf a; pack8(a, 0, x0, x1); pack8(a, 8, x2, x3);
    c0 = bf16_wmma(a, b0, c0);
    c1 = bf16_wmma(a, b1, c1);
  }
  const float bb0 = bout[col0 + laneN];
  const float bb1 = bout[col0 + 16 + laneN];
#pragma unroll
  for (int v = 0; v < 8; ++v) {
    int m = v + (hi << 3);
    size_t base = (size_t)(row0 + m) * INF_ + laneN;
    out[base + col0]      = c0[v] + bb0;
    out[base + col0 + 16] = c1[v] + bb1;
  }
}

// ---------------- attention / edge kernels ----------------

__global__ void __launch_bounds__(256)
k_attn(const __bf16* __restrict__ hbf, const float* __restrict__ attn_l,
       const float* __restrict__ attn_r, float* __restrict__ el,
       float* __restrict__ er) {
  int wave = (int)((blockIdx.x * blockDim.x + threadIdx.x) >> 5);
  int lane = threadIdx.x & 31;
  if (wave >= NN * HH) return;
  int n = wave / HH, h = wave % HH;
  const __bf16* hp = hbf + (size_t)n * HDIM + h * DD;
  const float* al = attn_l + h * DD;
  const float* ar = attn_r + h * DD;
  float sl = 0.f, sr = 0.f;
#pragma unroll
  for (int j = 0; j < DD; j += 32) {
    float hv = (float)hp[j + lane];
    sl += hv * al[j + lane];
    sr += hv * ar[j + lane];
  }
  sl = warp_reduce_sum(sl);
  sr = warp_reduce_sum(sr);
  if (lane == 0) { el[wave] = sl; er[wave] = sr; }
}

__global__ void k_edge_max(const int* __restrict__ src, const int* __restrict__ dst,
                           const float* __restrict__ el, const float* __restrict__ er,
                           float* __restrict__ emax) {
  int i = blockIdx.x * blockDim.x + threadIdx.x;
  if (i >= EE * HH) return;
  int e = i >> 2, h = i & 3;
  int d = dst[e];
  float v = edge_score(el, er, src[e], d, h);
  atomic_max_float(&emax[d * HH + h], v);
}

__global__ void k_edge_sum(const int* __restrict__ src, const int* __restrict__ dst,
                           const float* __restrict__ el, const float* __restrict__ er,
                           const float* __restrict__ emax, float* __restrict__ denom) {
  int i = blockIdx.x * blockDim.x + threadIdx.x;
  if (i >= EE * HH) return;
  int e = i >> 2, h = i & 3;
  int d = dst[e];
  float v = edge_score(el, er, src[e], d, h);
  atomicAdd(&denom[d * HH + h], __expf(v - emax[d * HH + h]));
}

// rst[dst] += a * h[src]; 128 threads per edge (one per d), 4 heads each
__global__ void __launch_bounds__(256)
k_edge_scatter(const int* __restrict__ src, const int* __restrict__ dst,
               const float* __restrict__ el, const float* __restrict__ er,
               const float* __restrict__ emax, const float* __restrict__ denom,
               const __bf16* __restrict__ hbf, float* __restrict__ rst) {
  int t = threadIdx.x;
  int e = blockIdx.x * 2 + (t >> 7);
  int d = t & 127;
  if (e >= EE) return;
  int s = src[e], dn = dst[e];
  float a[HH];
#pragma unroll
  for (int h = 0; h < HH; ++h) {
    float v = edge_score(el, er, s, dn, h);
    a[h] = __expf(v - emax[dn * HH + h]) / denom[dn * HH + h];
  }
  const __bf16* hp = hbf + (size_t)s * HDIM;
  float* rp = rst + (size_t)dn * HDIM;
#pragma unroll
  for (int h = 0; h < HH; ++h)
    atomicAdd(&rp[h * DD + d], (float)hp[h * DD + d] * a[h]);
}

// ---------------- BatchNorm ----------------

__global__ void __launch_bounds__(512)
k_bn_stats(const float* __restrict__ x, float* __restrict__ bnsum,
           float* __restrict__ bnsq) {
  int c = threadIdx.x;                 // blockDim = 512
  float s = 0.f, s2 = 0.f;
  for (int r = blockIdx.x; r < NN; r += gridDim.x) {
    float v = x[(size_t)r * HDIM + c];
    s += v; s2 += v * v;
  }
  atomicAdd(&bnsum[c], s);
  atomicAdd(&bnsq[c], s2);
}

__global__ void k_bn_finalize(const float* __restrict__ bnsum,
                              const float* __restrict__ bnsq,
                              const float* __restrict__ gamma,
                              const float* __restrict__ beta,
                              float* __restrict__ scale, float* __restrict__ shift) {
  int c = threadIdx.x;                 // 512 threads, 1 block
  float mu  = bnsum[c] / (float)NN;
  float var = bnsq[c] / (float)NN - mu * mu;
  float rstd = rsqrtf(var + BN_EPS);
  float sc = gamma[c] * rstd;
  scale[c] = sc;
  shift[c] = beta[c] - mu * sc;
}

// ---------------- launch ----------------

extern "C" void kernel_launch(void* const* d_in, const int* in_sizes, int n_in,
                              void* d_out, int out_size, void* d_ws, size_t ws_size,
                              hipStream_t stream) {
  (void)in_sizes; (void)n_in; (void)out_size; (void)ws_size;
  const float* feat  = (const float*)d_in[0];
  const int*   src   = (const int*)d_in[1];
  const int*   dst   = (const int*)d_in[2];
  const float* Wfc   = (const float*)d_in[3];
  const float* attnl = (const float*)d_in[4];
  const float* attnr = (const float*)d_in[5];
  const float* Wres  = (const float*)d_in[6];
  const float* biasg = (const float*)d_in[7];
  const float* gamma = (const float*)d_in[8];
  const float* beta  = (const float*)d_in[9];
  const float* Wout  = (const float*)d_in[10];
  const float* bout  = (const float*)d_in[11];
  float* out = (float*)d_out;

  char* ws = (char*)d_ws;
  size_t off = 0;
  auto carve = [&](size_t bytes) -> char* {
    char* p = ws + off;
    off += (bytes + 255) & ~(size_t)255;
    return p;
  };
  __bf16* hbf    = (__bf16*)carve((size_t)NN * HDIM * sizeof(__bf16)); // 30.7 MB
  float*  rst    = (float*)carve((size_t)NN * HDIM * sizeof(float));   // 61.4 MB
  __bf16* featbf = (__bf16*)carve((size_t)NN * INF_ * sizeof(__bf16)); // 7.7 MB
  __bf16* wtfc   = (__bf16*)carve((size_t)HDIM * INF_ * sizeof(__bf16));
  __bf16* wtres  = (__bf16*)carve((size_t)HDIM * INF_ * sizeof(__bf16));
  __bf16* wtout  = (__bf16*)carve((size_t)INF_ * HDIM * sizeof(__bf16));
  float* el    = (float*)carve((size_t)NN * HH * sizeof(float));
  float* er    = (float*)carve((size_t)NN * HH * sizeof(float));
  float* emax  = (float*)carve((size_t)NN * HH * sizeof(float));
  float* denom = (float*)carve((size_t)NN * HH * sizeof(float));
  float* bnsum = (float*)carve(HDIM * sizeof(float));
  float* bnsq  = (float*)carve(HDIM * sizeof(float));
  float* scale = (float*)carve(HDIM * sizeof(float));
  float* shift = (float*)carve(HDIM * sizeof(float));

  // prep: bf16 conversions / weight transposes (tiny, L2-resident afterwards)
  k_cvt_feat<<<(NN * INF_ / 4 + 255) / 256, 256, 0, stream>>>(feat, featbf);
  k_transpose_w<<<(INF_ * HDIM + 255) / 256, 256, 0, stream>>>(Wfc,  wtfc,  INF_, HDIM);
  k_transpose_w<<<(INF_ * HDIM + 255) / 256, 256, 0, stream>>>(Wres, wtres, INF_, HDIM);
  k_transpose_w<<<(HDIM * INF_ + 255) / 256, 256, 0, stream>>>(Wout, wtout, HDIM, INF_);
  k_init<<<(NN * HH + 255) / 256, 256, 0, stream>>>(emax, denom, bnsum, bnsq);

  // 1875 row tiles * 16 col-pairs = 30000 waves / 8 per block
  k_gemm_fc_res<<<3750, 256, 0, stream>>>(featbf, wtfc, wtres, biasg, hbf, rst);
  // 120000 (n,h) waves / 8 per block
  k_attn<<<15000, 256, 0, stream>>>(hbf, attnl, attnr, el, er);
  k_edge_max<<<(EE * HH + 255) / 256, 256, 0, stream>>>(src, dst, el, er, emax);
  k_edge_sum<<<(EE * HH + 255) / 256, 256, 0, stream>>>(src, dst, el, er, emax, denom);
  k_edge_scatter<<<EE / 2, 256, 0, stream>>>(src, dst, el, er, emax, denom, hbf, rst);
  k_bn_stats<<<120, 512, 0, stream>>>(rst, bnsum, bnsq);
  k_bn_finalize<<<1, 512, 0, stream>>>(bnsum, bnsq, gamma, beta, scale, shift);
  // 1875 row tiles * 4 col-pairs = 7500 waves / 8 per block -> 938 blocks
  k_gemm_out<<<938, 256, 0, stream>>>(rst, scale, shift, wtout, bout, out);
}
